// MinGRUCell_10393820856372
// MI455X (gfx1250) — matrix operand order repaired and checked by reading, not compiled, not run on hardware
//
#include <hip/hip_runtime.h>
#include <hip/hip_bf16.h>
#include <math.h>

// ---------------------------------------------------------------------------
// MinGRU forward for MI455X (gfx1250, wave32).
//  Phase 0: one-time f32 -> bf16 conversion of x and W into d_ws.
//  Phase 1: WMMA bf16 GEMM zh = x @ W^T + b with LDS double-buffered B tile
//           (B shared by all 8 waves of a block; staged once per block, read
//           via ds_load_b128), fused linear-space gate epilogue:
//              coeff = 1 - sigmoid(z)        -> d_ws
//              value = sigmoid(z) * g(h~)    -> d_out
//  Phase 2: sequential per-(n,h) recurrence h = coeff*h + value: one dependent
//           FMA per step, 2 loads + 1 store, pure HBM-roofline stream.
// ---------------------------------------------------------------------------

typedef __attribute__((ext_vector_type(16))) __bf16 v16bf;
typedef __attribute__((ext_vector_type(8)))  __bf16 v8bf;
typedef __attribute__((ext_vector_type(4)))  __bf16 v4bf;
typedef __attribute__((ext_vector_type(8)))  float  v8f;

#define LDS_ROW 40   // bf16 elems per LDS row (32 data + 8 pad = 80B, 16B-aligned)

// ---------------- Phase 0: f32 -> bf16 conversion (vectorized) --------------
__global__ void __launch_bounds__(256)
cvt_f32_bf16_kernel(const float* __restrict__ in, __bf16* __restrict__ out, int n4) {
    int i = blockIdx.x * blockDim.x + threadIdx.x;
    if (i >= n4) return;
    float4 f = reinterpret_cast<const float4*>(in)[i];
    v4bf o;
    o[0] = (__bf16)f.x; o[1] = (__bf16)f.y; o[2] = (__bf16)f.z; o[3] = (__bf16)f.w;
    reinterpret_cast<v4bf*>(out)[i] = o;
}

// ---------------- Phase 1: WMMA GEMM + fused gate epilogue ------------------
// A fragment (16x32 MxK bf16): lane&15 = row; per-lane K run = klo+{0..15},
// klo = (lane>>4)*16  -> two contiguous 16B bf16 loads.
__device__ __forceinline__ v16bf load_a_frag(const __bf16* __restrict__ row, int kb, int klo) {
    v8bf lo = *reinterpret_cast<const v8bf*>(row + kb + klo);
    v8bf hi = *reinterpret_cast<const v8bf*>(row + kb + klo + 8);
    return __builtin_shufflevector(lo, hi, 0,1,2,3,4,5,6,7,8,9,10,11,12,13,14,15);
}

// B fragment (32x16 KxN bf16) from the staged LDS tile: lane&15 = column
// (a row of W); per-lane K run = (lane>>4)*16 + {0..15} -> 32B LDS read.
__device__ __forceinline__ v16bf lds_b_frag(const __bf16* __restrict__ buf, int grp,
                                            int lane15, int laneHalf) {
    return *reinterpret_cast<const v16bf*>(buf + (size_t)(grp * 16 + lane15) * LDS_ROW
                                               + laneHalf * 16);
}

__device__ __forceinline__ float sigmoid_f(float v) {
    return __fdividef(1.0f, 1.0f + __expf(-v));
}

// 256 threads = 8 wave32. Wave w owns m-tile (blockIdx.y*8+w)*16; the block
// owns a 32-column span: z cols [oT,oT+32) and h_inter cols [H+oT,H+oT+32).
// The 64-row B tile is shared by all waves -> staged in LDS, double buffered.
__global__ void __launch_bounds__(256)
mingru_gemm_gate_kernel(const __bf16* __restrict__ xbf,  // (M, K) bf16
                        const __bf16* __restrict__ wbf,  // (2H, K) bf16
                        const float*  __restrict__ b,    // (2H)
                        float* __restrict__ value,       // (M, H) -> d_out
                        float* __restrict__ coeff,       // (M, H) -> d_ws
                        int M, int K, int H) {
    __shared__ __bf16 ldsB[2][64 * LDS_ROW];   // 2 x 5KB

    const int tid  = threadIdx.x;
    const int wave = tid >> 5;
    const int lane = tid & 31;
    const int lane15   = lane & 15;
    const int laneHalf = lane >> 4;

    const int mTile = (blockIdx.y * 8 + wave) * 16;
    const int oTile = blockIdx.x * 32;

    // --- staging role of this thread: row r (0..63), 16B segment seg (0..3)
    const int r   = tid >> 2;
    const int seg = tid & 3;
    const int wrow = (r < 32) ? (oTile + r) : (H + oTile + (r - 32));
    const __bf16* __restrict__ stageSrc = wbf + (size_t)wrow * K + seg * 8;
    __bf16* const stageDst0 = &ldsB[0][(size_t)r * LDS_ROW + seg * 8];
    __bf16* const stageDst1 = &ldsB[1][(size_t)r * LDS_ROW + seg * 8];

    const int klo = laneHalf * 16;   // A-frag per-lane K offset
    const __bf16* __restrict__ xrow = xbf + (size_t)(mTile + lane15) * K;

    v8f acc_z0 = {}, acc_z1 = {}, acc_h0 = {}, acc_h1 = {};

    const int nk = K / 32;

    // preload tile 0 into buffer 0
    {
        v8bf s = *reinterpret_cast<const v8bf*>(stageSrc);
        *reinterpret_cast<v8bf*>(stageDst0) = s;
    }
    __syncthreads();

    for (int i = 0; i < nk; ++i) {
        const int cur = i & 1;
        const int kb  = i * 32;

        // issue next tile's global fetch early (latency overlap with WMMAs)
        v8bf nxt = {};
        if (i + 1 < nk) {
            nxt = *reinterpret_cast<const v8bf*>(stageSrc + kb + 32);
            __builtin_prefetch(xrow + kb + 32 + klo, 0, 3);
        }

        const __bf16* buf = ldsB[cur];
        v16bf a   = load_a_frag(xrow, kb, klo);
        v16bf bz0 = lds_b_frag(buf, 0, lane15, laneHalf);
        v16bf bz1 = lds_b_frag(buf, 1, lane15, laneHalf);
        v16bf bh0 = lds_b_frag(buf, 2, lane15, laneHalf);
        v16bf bh1 = lds_b_frag(buf, 3, lane15, laneHalf);

        acc_z0 = __builtin_amdgcn_wmma_f32_16x16x32_bf16(false, a, false, bz0, (short)0, acc_z0, false, false);
        acc_z1 = __builtin_amdgcn_wmma_f32_16x16x32_bf16(false, a, false, bz1, (short)0, acc_z1, false, false);
        acc_h0 = __builtin_amdgcn_wmma_f32_16x16x32_bf16(false, a, false, bh0, (short)0, acc_h0, false, false);
        acc_h1 = __builtin_amdgcn_wmma_f32_16x16x32_bf16(false, a, false, bh1, (short)0, acc_h1, false, false);

        if (i + 1 < nk) {
            // write the other buffer (not read this iteration), then publish
            *reinterpret_cast<v8bf*>(cur ? stageDst0 : stageDst1) = nxt;
            __syncthreads();
        }
    }

    // C/D layout: element rr -> row = mTile + rr + laneHalf*8, col = lane&15.
    const float bz0v = b[oTile + lane15];
    const float bz1v = b[oTile + 16 + lane15];
    const float bh0v = b[H + oTile + lane15];
    const float bh1v = b[H + oTile + 16 + lane15];
#pragma unroll
    for (int rr = 0; rr < 8; ++rr) {
        const size_t row = (size_t)(mTile + rr + laneHalf * 8);
        {
            const float z  = acc_z0[rr] + bz0v;
            const float hi = acc_h0[rr] + bh0v;
            const float sz = sigmoid_f(z);
            const float g  = (hi >= 0.0f) ? (hi + 0.5f) : sigmoid_f(hi);
            const size_t off = row * H + (oTile + lane15);
            coeff[off] = 1.0f - sz;
            value[off] = sz * g;
        }
        {
            const float z  = acc_z1[rr] + bz1v;
            const float hi = acc_h1[rr] + bh1v;
            const float sz = sigmoid_f(z);
            const float g  = (hi >= 0.0f) ? (hi + 0.5f) : sigmoid_f(hi);
            const size_t off = row * H + (oTile + 16 + lane15);
            coeff[off] = 1.0f - sz;
            value[off] = sz * g;
        }
    }
}

// ---------------- Phase 2: linear recurrence scan ---------------------------
// One thread per (n,h) channel; lane-consecutive h => fully coalesced 128B
// accesses per wave per step. Dependent chain: one FMA per step.
__global__ void __launch_bounds__(256)
mingru_scan_kernel(const float* __restrict__ hx,     // (N, H)
                   float* __restrict__ io_value,     // (N, L, H): value in, h out
                   const float* __restrict__ coeff,  // (N, L, H)
                   int N, int L, int H) {
    const int idx = blockIdx.x * blockDim.x + threadIdx.x;
    if (idx >= N * H) return;
    const int n = idx / H;
    const int h = idx - n * H;

    float hs = hx[idx];
    const size_t base = (size_t)n * L * H + h;
    for (int l = 0; l < L; ++l) {
        const size_t off = base + (size_t)l * H;
        const float v = io_value[off];
        const float c = coeff[off];
        hs = fmaf(c, hs, v);          // h = (1-z)*h + z*g
        io_value[off] = hs;
    }
}

extern "C" void kernel_launch(void* const* d_in, const int* in_sizes, int n_in,
                              void* d_out, int out_size, void* d_ws, size_t ws_size,
                              hipStream_t stream) {
    const float* x  = (const float*)d_in[0];  // (N, L, H_in) f32
    const float* W  = (const float*)d_in[1];  // (2H, H_in)   f32
    const float* b  = (const float*)d_in[2];  // (2H,)        f32
    const float* hx = (const float*)d_in[3];  // (N, H)       f32
    float* out = (float*)d_out;               // (N, L, H)    f32

    const int H2 = in_sizes[2];               // 2H = 2048
    const int H  = H2 / 2;                    // 1024
    const int K  = in_sizes[1] / H2;          // H_in = 1024
    const int N  = in_sizes[3] / H;           // 8
    const int M  = in_sizes[0] / K;           // N*L = 32768
    const int L  = M / N;                     // 4096

    // Workspace layout: [coeff f32 (M*H)] [xbf16 (M*K)] [Wbf16 (2H*K)]
    float*  coeff = (float*)d_ws;
    __bf16* xbf   = (__bf16*)((char*)d_ws + (size_t)M * H * sizeof(float));
    __bf16* wbf   = xbf + (size_t)M * K;

    // Phase 0: one-time bf16 conversion.
    {
        const int nx4 = (M * K) / 4;
        cvt_f32_bf16_kernel<<<(nx4 + 255) / 256, 256, 0, stream>>>(x, xbf, nx4);
        const int nw4 = (H2 * K) / 4;
        cvt_f32_bf16_kernel<<<(nw4 + 255) / 256, 256, 0, stream>>>(W, wbf, nw4);
    }

    // Phase 1: WMMA GEMM (LDS double-buffered B) + fused gate math.
    {
        dim3 gblock(256);
        dim3 ggrid(H / 32, M / 128);          // 8 wave-tiles of 16 rows, 32 cols
        mingru_gemm_gate_kernel<<<ggrid, gblock, 0, stream>>>(xbf, wbf, b, out, coeff, M, K, H);
    }

    // Phase 2: recurrence scan (one FMA per step per channel).
    {
        const int nch = N * H;
        mingru_scan_kernel<<<(nch + 255) / 256, 256, 0, stream>>>(hx, out, coeff, N, L, H);
    }
}
